// AutoDeepFM_30502857736455
// MI455X (gfx1250) — matrix operand order repaired and checked by reading, compiled, not run on hardware
//
#include <hip/hip_runtime.h>
#include <hip/hip_bf16.h>

// ---------------- types for WMMA ----------------
typedef __attribute__((ext_vector_type(16))) _Float16 v16h;
typedef __attribute__((ext_vector_type(8)))  _Float16 v8h;
typedef __attribute__((ext_vector_type(8)))  float    v8f;

union V16H { v16h v; v8h h[2]; };

// ---------------- problem constants ----------------
#define BATCH 512
#define NFIELD 39
#define EDIM 16
#define FE 624           // NFIELD * EDIM
#define KPAD1 640        // FE padded to multiple of 32
#define HID 700
#define HPAD 704         // HID padded to multiple of 16 (and 32 for K)
#define NPAIR 741
#define NTRIP 9139

// =====================================================================
// Kernel 0: generate combination index tables (matches itertools order)
// =====================================================================
__global__ void gen_idx_kernel(int* __restrict__ pairs, int* __restrict__ trips) {
    if (blockIdx.x == 0 && threadIdx.x == 0) {
        int p = 0;
        for (int i = 0; i < NFIELD; ++i)
            for (int j = i + 1; j < NFIELD; ++j) {
                pairs[2 * p] = i; pairs[2 * p + 1] = j; ++p;
            }
        int t = 0;
        for (int i = 0; i < NFIELD; ++i)
            for (int j = i + 1; j < NFIELD; ++j)
                for (int k = j + 1; k < NFIELD; ++k) {
                    trips[3 * t] = i; trips[3 * t + 1] = j; trips[3 * t + 2] = k; ++t;
                }
    }
}

// =====================================================================
// Kernel 1: convert a (Nreal x Kreal) f32 weight to (Npad x Kpad) f16,
//           zero padded.
// =====================================================================
__global__ void convert_pad_kernel(const float* __restrict__ src,
                                   _Float16* __restrict__ dst,
                                   int Nreal, int Kreal, int Kpad, int total) {
    int idx = blockIdx.x * blockDim.x + threadIdx.x;
    if (idx >= total) return;
    int n = idx / Kpad;
    int k = idx - n * Kpad;
    float v = 0.0f;
    if (n < Nreal && k < Kreal) v = src[(size_t)n * Kreal + k];
    dst[idx] = (_Float16)v;
}

// =====================================================================
// Kernel 2: embedding gathers.  Builds f16 activation matrix A0
//           (BATCH x KPAD1) and f32 copies of xv / xps (BATCH x FE).
// =====================================================================
__global__ void gather_kernel(const int* __restrict__ inputs,
                              const float* __restrict__ Ev,
                              const float* __restrict__ Eps,
                              _Float16* __restrict__ A0,
                              float* __restrict__ xvf,
                              float* __restrict__ xpsf) {
    int idx = blockIdx.x * blockDim.x + threadIdx.x;
    if (idx >= BATCH * KPAD1) return;
    int b = idx / KPAD1;
    int col = idx - b * KPAD1;
    if (col < FE) {
        int f = col >> 4;
        int e = col & 15;
        int row = inputs[b * NFIELD + f];
        float v = Ev[(size_t)row * EDIM + e];
        float p = Eps[(size_t)row * EDIM + e];
        A0[idx] = (_Float16)v;
        xvf[b * FE + col] = v;
        xpsf[b * FE + col] = p;
    } else {
        A0[idx] = (_Float16)0.0f;
    }
}

// =====================================================================
// Kernel 3: WMMA GEMM.  D = act(A @ W^T + bias), f16 in / f32 acc.
//   A : M x Kpad row-major f16   (M = BATCH)
//   W : Npad x Kpad row-major f16 (so B[k][n] = W[n][k])
//   D : M x Npad row-major f16
// One wave (32 threads) computes one 16x16 tile of D.
// Fragment layouts per CDNA5 ISA 7.12.2 (wave32):
//   A 16x32 f16 : lanes 0-15 -> M=lane; halves 0..7 -> K = k0+hi*8+j,
//                 halves 8..15 -> K = k0+16+hi*8+(j-8)     (hi = lane>>4)
//   B 32x16 f16 : column N = lane&15; same K mapping as A
//   C/D 16x16 f32: VGPR i -> row M = hi*8+i, col N = lane&15
// =====================================================================
__global__ void __launch_bounds__(32)
gemm_wmma_kernel(const _Float16* __restrict__ A,
                 const _Float16* __restrict__ W,
                 const float* __restrict__ bias,
                 _Float16* __restrict__ D,
                 int Kpad, int Npad, int Nreal, int relu) {
    const int lane = threadIdx.x;
    const int r    = lane & 15;
    const int hi   = lane >> 4;
    const int mbase = blockIdx.y << 4;
    const int nbase = blockIdx.x << 4;

    const _Float16* __restrict__ Arow = A + (size_t)(mbase + r) * Kpad;
    const _Float16* __restrict__ Wrow = W + (size_t)(nbase + r) * Kpad;

    v8f acc = {};
    for (int k = 0; k < Kpad; k += 32) {
        V16H a, b;
        a.h[0] = *(const v8h*)(Arow + k + hi * 8);
        a.h[1] = *(const v8h*)(Arow + k + 16 + hi * 8);
        b.h[0] = *(const v8h*)(Wrow + k + hi * 8);
        b.h[1] = *(const v8h*)(Wrow + k + 16 + hi * 8);
        acc = __builtin_amdgcn_wmma_f32_16x16x32_f16(
            /*neg_a=*/false, a.v, /*neg_b=*/false, b.v,
            /*c_mod=*/(short)0, acc, /*reuse_a=*/false, /*reuse_b=*/false);
    }

    const int n = nbase + r;
    const float bn = (n < Nreal) ? bias[n] : 0.0f;
#pragma unroll
    for (int i = 0; i < 8; ++i) {
        int m = mbase + hi * 8 + i;
        float v = acc[i] + bn;
        if (relu) v = fmaxf(v, 0.0f);
        D[(size_t)m * Npad + n] = (_Float16)v;
    }
}

// =====================================================================
// Kernel 4: fused linear + FM-pairwise + weighted-triple terms.
// One block (256 threads) per batch row; embeddings staged in LDS.
// =====================================================================
__global__ void __launch_bounds__(256)
interactions_kernel(const int* __restrict__ inputs,
                    const float* __restrict__ Ww,   // (E,F)
                    const float* __restrict__ bw,   // (E)
                    const float* __restrict__ Wl,   // (1,E)
                    const float* __restrict__ bl,   // (1)
                    const float* __restrict__ xvf,  // (B,FE)
                    const float* __restrict__ xpsf, // (B,FE)
                    const float* __restrict__ edge_w,
                    const float* __restrict__ bn_g,
                    const float* __restrict__ bn_b,
                    const float* __restrict__ bn_m,
                    const float* __restrict__ bn_v,
                    const float* __restrict__ w3,
                    const int* __restrict__ pairs,
                    const int* __restrict__ trips,
                    float* __restrict__ misc) {
    __shared__ float xv_s[FE];
    __shared__ float xp_s[FE];
    __shared__ float red[256];

    const int b = blockIdx.x;
    const int t = threadIdx.x;

    for (int i = t; i < FE; i += 256) {
        xv_s[i] = xvf[b * FE + i];
        xp_s[i] = xpsf[b * FE + i];
    }
    __syncthreads();

    float acc = 0.0f;

    // ---- linear term:  l = sum_f x[b,f] * (sum_e Wl[e]*Ww[e,f])  + const
    if (t < NFIELD) {
        float coef = 0.0f;
#pragma unroll
        for (int e = 0; e < EDIM; ++e) coef += Wl[e] * Ww[e * NFIELD + t];
        acc += (float)inputs[b * NFIELD + t] * coef;
    }
    if (t == 0) {
        float c0 = bl[0];
#pragma unroll
        for (int e = 0; e < EDIM; ++e) c0 += Wl[e] * bw[e];
        acc += c0;
    }

    // ---- pairwise FM with BN-affine
    for (int p = t; p < NPAIR; p += 256) {
        int i = pairs[2 * p];
        int j = pairs[2 * p + 1];
        float d = 0.0f;
#pragma unroll
        for (int e = 0; e < EDIM; ++e) d += xv_s[i * EDIM + e] * xv_s[j * EDIM + e];
        float bn = (d - bn_m[p]) * rsqrtf(bn_v[p] + 1e-5f) * bn_g[p] + bn_b[p];
        acc += bn * edge_w[p];
    }

    // ---- weighted triples
    for (int q = t; q < NTRIP; q += 256) {
        int i = trips[3 * q];
        int j = trips[3 * q + 1];
        int k = trips[3 * q + 2];
        float d = 0.0f;
#pragma unroll
        for (int e = 0; e < EDIM; ++e)
            d += xp_s[i * EDIM + e] * xp_s[j * EDIM + e] * xp_s[k * EDIM + e];
        acc += d * w3[q];
    }

    red[t] = acc;
    __syncthreads();
#pragma unroll
    for (int s = 128; s > 0; s >>= 1) {
        if (t < s) red[t] += red[t + s];
        __syncthreads();
    }
    if (t == 0) misc[b] = red[0];
}

// =====================================================================
// Kernel 5: final 700-dot with W4 + combine all terms.
// =====================================================================
__global__ void finalize_kernel(const _Float16* __restrict__ H3,
                                const float* __restrict__ W4,
                                const float* __restrict__ b4,
                                const float* __restrict__ misc,
                                float* __restrict__ out) {
    int b = blockIdx.x * blockDim.x + threadIdx.x;
    if (b >= BATCH) return;
    float acc = b4[0];
    const _Float16* row = H3 + (size_t)b * HPAD;
    for (int n = 0; n < HID; ++n) acc += (float)row[n] * W4[n];
    out[b] = misc[b] + acc;
}

// =====================================================================
// launcher
// =====================================================================
static inline size_t align_up(size_t x) { return (x + 255) & ~(size_t)255; }

extern "C" void kernel_launch(void* const* d_in, const int* in_sizes, int n_in,
                              void* d_out, int out_size, void* d_ws, size_t ws_size,
                              hipStream_t stream) {
    (void)in_sizes; (void)n_in; (void)out_size; (void)ws_size;

    const int*   inputs = (const int*)  d_in[0];
    const float* Ww     = (const float*)d_in[1];
    const float* bw     = (const float*)d_in[2];
    const float* Wl     = (const float*)d_in[3];
    const float* bl     = (const float*)d_in[4];
    const float* Ev     = (const float*)d_in[5];
    const float* Eps    = (const float*)d_in[6];
    const float* W1     = (const float*)d_in[7];
    const float* b1     = (const float*)d_in[8];
    const float* W2     = (const float*)d_in[9];
    const float* b2     = (const float*)d_in[10];
    const float* W3     = (const float*)d_in[11];
    const float* b3     = (const float*)d_in[12];
    const float* W4     = (const float*)d_in[13];
    const float* b4     = (const float*)d_in[14];
    const float* edge_w = (const float*)d_in[15];
    const float* bn_g   = (const float*)d_in[16];
    const float* bn_b   = (const float*)d_in[17];
    const float* bn_m   = (const float*)d_in[18];
    const float* bn_v   = (const float*)d_in[19];
    const float* w3     = (const float*)d_in[20];
    float* out = (float*)d_out;

    // ---- workspace carve ----
    char* ws = (char*)d_ws;
    size_t off = 0;
    _Float16* W1h = (_Float16*)(ws + off); off = align_up(off + (size_t)HPAD * KPAD1 * 2);
    _Float16* W2h = (_Float16*)(ws + off); off = align_up(off + (size_t)HPAD * HPAD * 2);
    _Float16* W3h = (_Float16*)(ws + off); off = align_up(off + (size_t)HPAD * HPAD * 2);
    _Float16* A0  = (_Float16*)(ws + off); off = align_up(off + (size_t)BATCH * KPAD1 * 2);
    _Float16* H1  = (_Float16*)(ws + off); off = align_up(off + (size_t)BATCH * HPAD * 2);
    _Float16* H2  = (_Float16*)(ws + off); off = align_up(off + (size_t)BATCH * HPAD * 2);
    _Float16* H3  = (_Float16*)(ws + off); off = align_up(off + (size_t)BATCH * HPAD * 2);
    float*    xvf = (float*)   (ws + off); off = align_up(off + (size_t)BATCH * FE * 4);
    float*    xpf = (float*)   (ws + off); off = align_up(off + (size_t)BATCH * FE * 4);
    float*    msc = (float*)   (ws + off); off = align_up(off + (size_t)BATCH * 4);
    int*      prs = (int*)     (ws + off); off = align_up(off + (size_t)NPAIR * 2 * 4);
    int*      trs = (int*)     (ws + off); off = align_up(off + (size_t)NTRIP * 3 * 4);

    // 0) index tables
    gen_idx_kernel<<<1, 1, 0, stream>>>(prs, trs);

    // 1) weight conversion (f32 -> padded f16)
    {
        int tot1 = HPAD * KPAD1;
        convert_pad_kernel<<<(tot1 + 255) / 256, 256, 0, stream>>>(W1, W1h, HID, FE, KPAD1, tot1);
        int tot2 = HPAD * HPAD;
        convert_pad_kernel<<<(tot2 + 255) / 256, 256, 0, stream>>>(W2, W2h, HID, HID, HPAD, tot2);
        convert_pad_kernel<<<(tot2 + 255) / 256, 256, 0, stream>>>(W3, W3h, HID, HID, HPAD, tot2);
    }

    // 2) embedding gather
    {
        int tot = BATCH * KPAD1;
        gather_kernel<<<(tot + 255) / 256, 256, 0, stream>>>(inputs, Ev, Eps, A0, xvf, xpf);
    }

    // 3) fused linear + pairwise + triples
    interactions_kernel<<<BATCH, 256, 0, stream>>>(inputs, Ww, bw, Wl, bl, xvf, xpf,
                                                   edge_w, bn_g, bn_b, bn_m, bn_v, w3,
                                                   prs, trs, msc);

    // 4) MLP via WMMA GEMMs (one wave per 16x16 output tile)
    {
        dim3 blk(32, 1, 1);
        dim3 grd1(HPAD / 16, BATCH / 16, 1);
        gemm_wmma_kernel<<<grd1, blk, 0, stream>>>(A0, W1h, b1, H1, KPAD1, HPAD, HID, 1);
        gemm_wmma_kernel<<<grd1, blk, 0, stream>>>(H1, W2h, b2, H2, HPAD, HPAD, HID, 1);
        gemm_wmma_kernel<<<grd1, blk, 0, stream>>>(H2, W3h, b3, H3, HPAD, HPAD, HID, 1);
    }

    // 5) finalize
    finalize_kernel<<<(BATCH + 255) / 256, 256, 0, stream>>>(H3, W4, b4, msc, out);
}
